// CLOSpreadModel_43817256354311
// MI455X (gfx1250) — compile-verified, matrix operand order checked by reading, not run on hardware
//
#include <hip/hip_runtime.h>

// ---------------------------------------------------------------------------
// CLO spread model: out = base_hinge(mvoc) + adj_hinge[bucket](mvoc)
//                       + hinge(lev) + hinge(wap) + hinge(cpn) + hinge(nav) + biases
// Per wave: 16 rows/tile. Each hinge builds a 16x32 packed-f16 A-tile
// A[i,k] = relu(x_i - t_k) * w_k (pk_f16 VALU), then v_wmma_f32_16x16x32_f16
// with an all-ones B reduces over the 32 knots. Two independent f32 C
// accumulators break the WMMA->WMMA RAW chain so pk-f16 tile building
// co-executes in the hazard slots instead of v_nops.
// ---------------------------------------------------------------------------

typedef _Float16 h2   __attribute__((ext_vector_type(2)));
typedef _Float16 v16h __attribute__((ext_vector_type(16)));
typedef float    v8f  __attribute__((ext_vector_type(8)));

union APack { h2 h[8]; v16h v; };

__device__ __forceinline__ h2 relu2(h2 d) {
    h2 z = {(_Float16)0.0f, (_Float16)0.0f};
    return __builtin_elementwise_max(d, z);
}

// Pack 32 f32 params into 8 h2 regs in WMMA A-operand K order for this lane.
// koff = 0 for lanes 0-15 (K 0..7,16..23), 8 for lanes 16-31 (K 8..15,24..31).
__device__ __forceinline__ void pack8(const float* __restrict__ src, int koff, h2* dst) {
    float4 a0 = *(const float4*)(src + koff);
    float4 a1 = *(const float4*)(src + koff + 4);
    float4 b0 = *(const float4*)(src + 16 + koff);
    float4 b1 = *(const float4*)(src + 16 + koff + 4);
    dst[0] = h2{(_Float16)a0.x, (_Float16)a0.y};
    dst[1] = h2{(_Float16)a0.z, (_Float16)a0.w};
    dst[2] = h2{(_Float16)a1.x, (_Float16)a1.y};
    dst[3] = h2{(_Float16)a1.z, (_Float16)a1.w};
    dst[4] = h2{(_Float16)b0.x, (_Float16)b0.y};
    dst[5] = h2{(_Float16)b0.z, (_Float16)b0.w};
    dst[6] = h2{(_Float16)b1.x, (_Float16)b1.y};
    dst[7] = h2{(_Float16)b1.z, (_Float16)b1.w};
}

struct Params {
    const float* mvoc; const int* bucket; const float* lev; const float* wap;
    const float* cpn;  const float* nav;
    const float* base_knots; const float* base_w; const float* base_b;
    const float* adj_knots;  const float* adj_w;  const float* adj_b;
    const float* idx_knots;  const float* idx_w;  const float* idx_b;
    const float* wap_knots;  const float* wap_w;  const float* wap_b;
    const float* cpn_knots;  const float* cpn_w;  const float* cpn_b;
    const float* nav_knots;  const float* nav_w;  const float* nav_b;
    const float* bias;
    float* out; int n;
};

__global__ __launch_bounds__(256) void clo_spread_wmma(Params p) {
    // Stage the 16 bucket experts (knots + weights) as f16 in LDS: 2 KB.
    __shared__ _Float16 sK[16 * 32];
    __shared__ _Float16 sW[16 * 32];
    for (int i = threadIdx.x; i < 16 * 32; i += blockDim.x) {
        sK[i] = (_Float16)p.adj_knots[i];
        sW[i] = (_Float16)p.adj_w[i];
    }
    __syncthreads();

    const int lane = threadIdx.x & 31;
    const int koff = (lane & 16) ? 8 : 0;   // K-half this lane owns in A layout

    // Lane-resident packed params for the 5 shared hinges (A-operand K order).
    h2 kn[5][8], wn[5][8];
    {
        const float* kp[5] = {p.base_knots, p.idx_knots, p.wap_knots, p.cpn_knots, p.nav_knots};
        const float* wp[5] = {p.base_w,     p.idx_w,     p.wap_w,     p.cpn_w,     p.nav_w};
#pragma unroll
        for (int h = 0; h < 5; ++h) { pack8(kp[h], koff, kn[h]); pack8(wp[h], koff, wn[h]); }
    }

    // All-ones B matrix: any layout of a constant matrix is the same splat.
    v16h bOnes;
#pragma unroll
    for (int i = 0; i < 16; ++i) bOnes[i] = (_Float16)1.0f;

    const float biasShared = p.base_b[0] + p.idx_b[0] + p.wap_b[0] +
                             p.cpn_b[0] + p.nav_b[0] + p.bias[0];

    const int tiles  = p.n >> 4;                                   // 16 rows per tile
    const int waveId = blockIdx.x * (blockDim.x >> 5) + (threadIdx.x >> 5);
    const int nWaves = gridDim.x * (blockDim.x >> 5);

    for (int t = waveId; t < tiles; t += nWaves) {
        const int rowBase = t << 4;
        const int r = rowBase + (lane & 15);   // lanes m and m+16 share row m

        float xm = p.mvoc[r];
        float xl = p.lev[r];
        float xw = p.wap[r];
        float xc = p.cpn[r];
        float xn = p.nav[r];
        int   b  = p.bucket[r];

        // Prefetch next tile's streams (global_prefetch_b8).
        {
            int rn = r + (nWaves << 4);
            if (rn < p.n) {
                __builtin_prefetch(p.mvoc + rn, 0, 0);
                __builtin_prefetch(p.lev  + rn, 0, 0);
                __builtin_prefetch(p.wap  + rn, 0, 0);
                __builtin_prefetch(p.cpn  + rn, 0, 0);
                __builtin_prefetch(p.nav  + rn, 0, 0);
                __builtin_prefetch(p.bucket + rn, 0, 0);
            }
        }

        // Two independent accumulator chains to break WMMA->WMMA RAW hazards.
        v8f acc0 = {};
        v8f acc1 = {};
        APack A;

        // ---- hinge 0: base(mvoc) -> acc0 ----
        h2 xs = {(_Float16)xm, (_Float16)xm};
#pragma unroll
        for (int j = 0; j < 8; ++j)
            A.h[j] = relu2(xs - kn[0][j]) * wn[0][j];
        acc0 = __builtin_amdgcn_wmma_f32_16x16x32_f16(false, A.v, false, bOnes,
                                                      (short)0, acc0, false, false);

        // ---- hinge 1: per-bucket adjustment on mvoc (LDS gather) -> acc1 ----
        {
            const h2* pk0 = (const h2*)(sK + b * 32 + koff);
            const h2* pk1 = (const h2*)(sK + b * 32 + 16 + koff);
            const h2* pw0 = (const h2*)(sW + b * 32 + koff);
            const h2* pw1 = (const h2*)(sW + b * 32 + 16 + koff);
#pragma unroll
            for (int j = 0; j < 4; ++j) {
                A.h[j]     = relu2(xs - pk0[j]) * pw0[j];
                A.h[4 + j] = relu2(xs - pk1[j]) * pw1[j];
            }
            acc1 = __builtin_amdgcn_wmma_f32_16x16x32_f16(false, A.v, false, bOnes,
                                                          (short)0, acc1, false, false);
        }

        // ---- hinges 2..5: lev / wap / cpn / nav, alternating accumulators ----
        const float xf[4] = {xl, xw, xc, xn};
#pragma unroll
        for (int h = 0; h < 4; ++h) {
            h2 s = {(_Float16)xf[h], (_Float16)xf[h]};
#pragma unroll
            for (int j = 0; j < 8; ++j)
                A.h[j] = relu2(s - kn[h + 1][j]) * wn[h + 1][j];
            if (h & 1)
                acc1 = __builtin_amdgcn_wmma_f32_16x16x32_f16(false, A.v, false, bOnes,
                                                              (short)0, acc1, false, false);
            else
                acc0 = __builtin_amdgcn_wmma_f32_16x16x32_f16(false, A.v, false, bOnes,
                                                              (short)0, acc0, false, false);
        }

        // Merge the two chains (8x v_add_f32).
        v8f acc = acc0 + acc1;

        // ---- extract: D row-sum is replicated across columns; pick c[lane&7],
        //      then gather across lane halves so lanes 0-15 hold rows 0-15.
        float v01 = (lane & 1) ? acc[1] : acc[0];
        float v23 = (lane & 1) ? acc[3] : acc[2];
        float v45 = (lane & 1) ? acc[5] : acc[4];
        float v67 = (lane & 1) ? acc[7] : acc[6];
        float vlo = (lane & 2) ? v23 : v01;
        float vhi = (lane & 2) ? v67 : v45;
        float v   = (lane & 4) ? vhi : vlo;
        int srcLane = ((lane & 8) ? 16 : 0) + (lane & 7);
        v = __int_as_float(
            __builtin_amdgcn_ds_bpermute(srcLane << 2, __float_as_int(v)));

        if (lane < 16) {
            p.out[r] = v + biasShared + p.adj_b[b];
        }
    }
}

extern "C" void kernel_launch(void* const* d_in, const int* in_sizes, int n_in,
                              void* d_out, int out_size, void* d_ws, size_t ws_size,
                              hipStream_t stream) {
    (void)in_sizes; (void)n_in; (void)d_ws; (void)ws_size;
    Params p;
    p.mvoc   = (const float*)d_in[0];
    p.bucket = (const int*)  d_in[1];
    p.lev    = (const float*)d_in[2];
    p.wap    = (const float*)d_in[3];
    p.cpn    = (const float*)d_in[4];
    p.nav    = (const float*)d_in[5];
    p.base_knots = (const float*)d_in[6];
    p.base_w     = (const float*)d_in[7];
    p.base_b     = (const float*)d_in[8];
    p.adj_knots  = (const float*)d_in[9];
    p.adj_w      = (const float*)d_in[10];
    p.adj_b      = (const float*)d_in[11];
    p.idx_knots  = (const float*)d_in[12];
    p.idx_w      = (const float*)d_in[13];
    p.idx_b      = (const float*)d_in[14];
    p.wap_knots  = (const float*)d_in[15];
    p.wap_w      = (const float*)d_in[16];
    p.wap_b      = (const float*)d_in[17];
    p.cpn_knots  = (const float*)d_in[18];
    p.cpn_w      = (const float*)d_in[19];
    p.cpn_b      = (const float*)d_in[20];
    p.nav_knots  = (const float*)d_in[21];
    p.nav_w      = (const float*)d_in[22];
    p.nav_b      = (const float*)d_in[23];
    p.bias       = (const float*)d_in[24];
    p.out = (float*)d_out;
    p.n   = out_size;   // 2,097,152 rows

    dim3 grid(2048), block(256);   // 16384 waves * 8 tiles of 16 rows = 2^21 rows
    hipLaunchKernelGGL(clo_spread_wmma, grid, block, 0, stream, p);
}
